// InGramEntityLayer_64046552318125
// MI455X (gfx1250) — matrix operand order repaired and checked by reading, not compiled, not run on hardware
//
#include <hip/hip_runtime.h>
#include <hip/hip_bf16.h>
#include <math.h>

// ---------------------------------------------------------------------------
// InGram entity layer for MI455X (gfx1250, wave32, WMMA 16x16x32 f16)
// ---------------------------------------------------------------------------

typedef _Float16 v16h __attribute__((ext_vector_type(16)));
typedef _Float16 v8h  __attribute__((ext_vector_type(8)));
typedef float    v8f  __attribute__((ext_vector_type(8)));

#define NDIM 64        // DIN = DREL = DOUT = 64
#define KATT 192       // 2*DIN + DREL
#define KAGG 128       // DIN + DREL
#define HEADS 8
#define EPSV 1e-6f
#define WPB 4          // waves (tiles) per block in WMMA kernels

#if defined(__gfx1250__) && __has_builtin(__builtin_amdgcn_global_load_async_to_lds_b128) && \
    __has_builtin(__builtin_amdgcn_s_wait_asynccnt)
#define HAVE_ASYNC_LDS 1
#endif

// Stage n16 * 16 bytes from global to LDS, cooperatively across the block.
__device__ __forceinline__ void stage_to_lds_16B(void* dst, const void* src, int n16,
                                                 int tid, int nthreads) {
#if defined(HAVE_ASYNC_LDS)
    typedef int v4i_t __attribute__((ext_vector_type(4)));
    typedef __attribute__((address_space(1))) v4i_t* gptr_t;   // global int4*
    typedef __attribute__((address_space(3))) v4i_t* lptr_t;   // LDS int4*
    gptr_t s = (gptr_t)src;                                    // drop const, AS1
    lptr_t d = (lptr_t)(unsigned)(unsigned long long)dst;      // LDS offset, AS3
    for (int i = tid; i < n16; i += nthreads)
        __builtin_amdgcn_global_load_async_to_lds_b128(s + i, d + i, 0, 0);
    __builtin_amdgcn_s_wait_asynccnt(0);
#else
    const uint4* s = (const uint4*)src;
    uint4* d = (uint4*)dst;
    for (int i = tid; i < n16; i += nthreads) d[i] = s[i];
#endif
}

// ---- order-preserving float <-> uint encoding for atomic max ----
__device__ __forceinline__ unsigned enc_f32(float f) {
    unsigned u = __float_as_uint(f);
    return (u & 0x80000000u) ? ~u : (u | 0x80000000u);
}
__device__ __forceinline__ float dec_f32(unsigned e) {
    unsigned u = (e & 0x80000000u) ? (e & 0x7fffffffu) : ~e;
    return __uint_as_float(u);
}

// ---------------------------------------------------------------------------
__global__ void k_zero(float* p, long long n) {
    long long stride = (long long)gridDim.x * blockDim.x;
    for (long long i = (long long)blockIdx.x * blockDim.x + threadIdx.x; i < n; i += stride)
        p[i] = 0.0f;
}

__global__ void k_f32_to_f16(const float* __restrict__ s, _Float16* __restrict__ d, long long n) {
    long long stride = (long long)gridDim.x * blockDim.x;
    for (long long i = (long long)blockIdx.x * blockDim.x + threadIdx.x; i < n; i += stride)
        d[i] = (_Float16)s[i];
}

// in-degree + sum of incoming relation embeddings per node
__global__ void k_degree(const float* __restrict__ emb_rel,
                         const int* __restrict__ tail_idx,
                         const int* __restrict__ rel_idx,
                         float* __restrict__ selfrel_f,
                         float* __restrict__ freq, int E) {
    long long n = (long long)E * NDIM;
    long long stride = (long long)gridDim.x * blockDim.x;
    for (long long i = (long long)blockIdx.x * blockDim.x + threadIdx.x; i < n; i += stride) {
        int e = (int)(i >> 6);
        int d = (int)(i & 63);
        int t = tail_idx[e];
        int r = rel_idx[e];
        atomicAdd(selfrel_f + (long long)t * NDIM + d, emb_rel[(long long)r * NDIM + d]);
        if (d == 0) atomicAdd(freq + t, 1.0f);
    }
}

__global__ void k_selfrel_fin(const float* __restrict__ selfrel_f,
                              const float* __restrict__ freq,
                              _Float16* __restrict__ selfrelH, int N) {
    long long n = (long long)N * NDIM;
    long long stride = (long long)gridDim.x * blockDim.x;
    for (long long i = (long long)blockIdx.x * blockDim.x + threadIdx.x; i < n; i += stride)
        selfrelH[i] = (_Float16)(selfrel_f[i] / (freq[i >> 6] + EPSV));
}

// ---------------------------------------------------------------------------
// Pass 1: attention logits.  raw[M,8] and per-(node,head) running max.
// One wave = one 16-edge tile. D = leaky(A[16,192] x W^T[192,64] + b) . v
// ---------------------------------------------------------------------------
__global__ __launch_bounds__(128) void k_attn(
    const _Float16* __restrict__ entH,      // [N,64]
    const _Float16* __restrict__ relH,      // [R,64]
    const _Float16* __restrict__ selfrelH,  // [N,64]
    const _Float16* __restrict__ attnWH,    // [64,192]
    const float* __restrict__ attn_b,       // [64]
    const float* __restrict__ attn_vec,     // [64] = [H,DH]
    const int* __restrict__ head_idx,
    const int* __restrict__ tail_idx,
    const int* __restrict__ rel_idx,
    float* __restrict__ raw,                // [M,8]
    unsigned* __restrict__ mx_enc,          // [N,8]
    int E, int M)
{
    __shared__ __align__(32) _Float16 ldsW[64 * KATT];  // 24 KB
    stage_to_lds_16B(ldsW, attnWH, 64 * KATT / 8, threadIdx.x, blockDim.x);
    __syncthreads();

    int wave = threadIdx.x >> 5;
    int lane = threadIdx.x & 31;
    int tile = blockIdx.x * WPB + wave;
    int ntiles = (M + 15) >> 4;
    if (tile >= ntiles) return;
    int base = tile << 4;

    // per-lane row of the A matrix (edge) r = lane & 15
    int r = lane & 15;
    int er = base + r;
    if (er >= M) er = M - 1;
    int t_r = (er < E) ? tail_idx[er] : (er - E);
    int h_r = (er < E) ? head_idx[er] : (er - E);
    const _Float16* rel_row = (er < E) ? (relH + (long long)rel_idx[er] * NDIM)
                                       : (selfrelH + (long long)(er - E) * NDIM);
    const _Float16* entT_row = entH + (long long)t_r * NDIM;
    const _Float16* entS_row = entH + (long long)h_r * NDIM;
    int koff = (lane >> 4) << 4;  // lanes 0-15 -> K 0..15 ; lanes 16-31 -> K 16..31

    // Load all 6 K-fragments of A (contiguous k = 16h + s per lane half, same order as B)
    v16h afr[6];
#pragma unroll
    for (int kb = 0; kb < 6; ++kb) {
        const _Float16* seg = (kb < 2) ? entT_row : (kb < 4) ? entS_row : rel_row;
        int cb = (kb & 1) << 5;
        v8h lo = *(const v8h*)(seg + cb + koff);
        v8h hi = *(const v8h*)(seg + cb + koff + 8);
        v16h a;
#pragma unroll
        for (int i = 0; i < 8; ++i) { a[i] = lo[i]; a[i + 8] = hi[i]; }
        afr[kb] = a;
    }

    int wrow = lane & 15;            // B column = output dim within tile
    int wcol = (lane >> 4) << 4;     // lanes 16-31 take K+16 half

#pragma unroll
    for (int nb = 0; nb < 4; ++nb) {
        v8f acc = {};
#pragma unroll
        for (int kb = 0; kb < 6; ++kb) {
            const v16h b = *(const v16h*)(&ldsW[(nb * 16 + wrow) * KATT + (kb << 5) + wcol]);
            acc = __builtin_amdgcn_wmma_f32_16x16x32_f16(
                false, afr[kb], false, b, (short)0, acc, false, false);
        }
        // epilogue: bias, leaky-relu(0.2), *attn_vec, reduce over dh (lanes ^1,^2,^4)
        int dim = nb * 16 + (lane & 15);
        float bias = attn_b[dim];
        float av = attn_vec[dim];
        int head = dim >> 3;
#pragma unroll
        for (int v = 0; v < 8; ++v) {
            int m = v + ((lane >> 4) << 3);      // edge row in tile
            float x = acc[v] + bias;
            x = (x > 0.0f) ? x : 0.2f * x;
            x *= av;
            x += __shfl_xor(x, 1, 32);
            x += __shfl_xor(x, 2, 32);
            x += __shfl_xor(x, 4, 32);
            int tm = __shfl(t_r, m, 32);         // tail of edge row m (all lanes active!)
            if ((lane & 7) == 0 && (base + m) < M) {
                raw[(long long)(base + m) * HEADS + head] = x;
                atomicMax(mx_enc + (long long)tm * HEADS + head, enc_f32(x));
            }
        }
    }
}

// ---------------------------------------------------------------------------
__global__ void k_exp(const int* __restrict__ tail_idx,
                      float* __restrict__ raw,
                      const unsigned* __restrict__ mx_enc,
                      float* __restrict__ sums, int E, int M) {
    long long n = (long long)M * HEADS;
    long long stride = (long long)gridDim.x * blockDim.x;
    for (long long i = (long long)blockIdx.x * blockDim.x + threadIdx.x; i < n; i += stride) {
        int e = (int)(i >> 3), h = (int)(i & 7);
        int t = (e < E) ? tail_idx[e] : (e - E);
        unsigned me = mx_enc[(long long)t * HEADS + h];
        float mx = me ? dec_f32(me) : 0.0f;     // empty segment -> 0 (matches reference)
        float a = __expf(raw[i] - mx);
        raw[i] = a;
        atomicAdd(sums + (long long)t * HEADS + h, a);
    }
}

// ---------------------------------------------------------------------------
// Pass 2: message projection + beta-weighted scatter-add.
// Beta normalization is fused into the staging of the per-tile beta block.
// ---------------------------------------------------------------------------
__global__ __launch_bounds__(128) void k_aggr(
    const _Float16* __restrict__ entH,
    const _Float16* __restrict__ relH,
    const _Float16* __restrict__ selfrelH,
    const _Float16* __restrict__ aggrWH,    // [64,128]
    const float* __restrict__ aggr_b,       // [64]
    const int* __restrict__ head_idx,
    const int* __restrict__ tail_idx,
    const int* __restrict__ rel_idx,
    const float* __restrict__ av,           // [M,8] un-normalized exp weights
    const float* __restrict__ sums,         // [N,8]
    float* __restrict__ out,                // [N,64]
    int E, int M)
{
    __shared__ __align__(32) _Float16 ldsW[64 * KAGG];   // 16 KB
    __shared__ float ldsBeta[WPB * 16 * HEADS];          // 2 KB
    stage_to_lds_16B(ldsW, aggrWH, 64 * KAGG / 8, threadIdx.x, blockDim.x);

    int wave = threadIdx.x >> 5;
    int lane = threadIdx.x & 31;
    int tile = blockIdx.x * WPB + wave;
    int ntiles = (M + 15) >> 4;
    bool valid = tile < ntiles;
    int base = valid ? (tile << 4) : 0;

    if (valid) {   // stage this wave's beta tile (16 edges x 8 heads), fused normalize
        long long lim = (long long)M * HEADS;
#pragma unroll
        for (int j = 0; j < 4; ++j) {
            long long bi = (long long)base * HEADS + lane * 4 + j;
            float b = 0.0f;
            if (bi < lim) {
                int e = (int)(bi >> 3), h = (int)(bi & 7);
                int t = (e < E) ? tail_idx[e] : (e - E);
                b = av[bi] / (sums[(long long)t * HEADS + h] + EPSV);
            }
            ldsBeta[wave * 128 + lane * 4 + j] = b;
        }
    }
    __syncthreads();
    if (!valid) return;

    int r = lane & 15;
    int er = base + r;
    if (er >= M) er = M - 1;
    int t_r = (er < E) ? tail_idx[er] : (er - E);
    int h_r = (er < E) ? head_idx[er] : (er - E);
    const _Float16* rel_row = (er < E) ? (relH + (long long)rel_idx[er] * NDIM)
                                       : (selfrelH + (long long)(er - E) * NDIM);
    const _Float16* entS_row = entH + (long long)h_r * NDIM;
    int koff = (lane >> 4) << 4;

    v16h afr[4];
#pragma unroll
    for (int kb = 0; kb < 4; ++kb) {
        const _Float16* seg = (kb < 2) ? entS_row : rel_row;
        int cb = (kb & 1) << 5;
        v8h lo = *(const v8h*)(seg + cb + koff);
        v8h hi = *(const v8h*)(seg + cb + koff + 8);
        v16h a;
#pragma unroll
        for (int i = 0; i < 8; ++i) { a[i] = lo[i]; a[i + 8] = hi[i]; }
        afr[kb] = a;
    }

    int wrow = lane & 15;
    int wcol = (lane >> 4) << 4;

#pragma unroll
    for (int nb = 0; nb < 4; ++nb) {
        v8f acc = {};
#pragma unroll
        for (int kb = 0; kb < 4; ++kb) {
            const v16h b = *(const v16h*)(&ldsW[(nb * 16 + wrow) * KAGG + (kb << 5) + wcol]);
            acc = __builtin_amdgcn_wmma_f32_16x16x32_f16(
                false, afr[kb], false, b, (short)0, acc, false, false);
        }
        int dim = nb * 16 + (lane & 15);
        float bias = aggr_b[dim];
        int head = dim >> 3;
#pragma unroll
        for (int v = 0; v < 8; ++v) {
            int m = v + ((lane >> 4) << 3);
            int tm = __shfl(t_r, m, 32);
            float g = acc[v] + bias;
            float bw = ldsBeta[wave * 128 + m * HEADS + head];
            if ((base + m) < M)
                atomicAdd(out + (long long)tm * NDIM + dim, bw * g);
        }
    }
}

// ---------------------------------------------------------------------------
extern "C" void kernel_launch(void* const* d_in, const int* in_sizes, int n_in,
                              void* d_out, int out_size, void* d_ws, size_t ws_size,
                              hipStream_t stream) {
    (void)n_in; (void)out_size; (void)ws_size;
    const float* emb_ent  = (const float*)d_in[0];
    const float* emb_rel  = (const float*)d_in[1];
    const float* attn_W   = (const float*)d_in[2];
    const float* attn_b   = (const float*)d_in[3];
    const float* attn_vec = (const float*)d_in[4];
    const float* aggr_W   = (const float*)d_in[5];
    const float* aggr_b   = (const float*)d_in[6];
    const int* head_idx   = (const int*)d_in[7];
    const int* tail_idx   = (const int*)d_in[8];
    const int* rel_idx    = (const int*)d_in[9];

    const int N = in_sizes[0] / NDIM;
    const int R = in_sizes[1] / NDIM;
    const int E = in_sizes[7];
    const int M = E + N;

    // ---- workspace bump allocator (256B aligned) ----
    char* ws = (char*)d_ws;
    size_t off = 0;
    auto alloc = [&](size_t bytes) -> char* {
        char* p = ws + off;
        off = (off + bytes + 255) & ~(size_t)255;
        return p;
    };
    float*     selfrel_f = (float*)alloc((size_t)N * NDIM * 4);
    float*     freq      = (float*)alloc((size_t)N * 4);
    float*     sums      = (float*)alloc((size_t)N * HEADS * 4);
    unsigned*  mx        = (unsigned*)alloc((size_t)N * HEADS * 4);
    float*     raw       = (float*)alloc((size_t)M * HEADS * 4);
    _Float16*  entH      = (_Float16*)alloc((size_t)N * NDIM * 2);
    _Float16*  relH      = (_Float16*)alloc((size_t)R * NDIM * 2);
    _Float16*  selfrelH  = (_Float16*)alloc((size_t)N * NDIM * 2);
    _Float16*  attnWH    = (_Float16*)alloc((size_t)64 * KATT * 2);
    _Float16*  aggrWH    = (_Float16*)alloc((size_t)64 * KAGG * 2);

    auto gs = [](long long n) {
        long long b = (n + 255) / 256;
        if (b > 4096) b = 4096;
        if (b < 1) b = 1;
        return (int)b;
    };

    // zero accumulators (+ output, which is scatter-add target)
    k_zero<<<gs((long long)N * NDIM), 256, 0, stream>>>(selfrel_f, (long long)N * NDIM);
    k_zero<<<gs(N), 256, 0, stream>>>(freq, N);
    k_zero<<<gs((long long)N * HEADS), 256, 0, stream>>>(sums, (long long)N * HEADS);
    k_zero<<<gs((long long)N * HEADS), 256, 0, stream>>>((float*)mx, (long long)N * HEADS);
    k_zero<<<gs((long long)N * NDIM), 256, 0, stream>>>((float*)d_out, (long long)N * NDIM);

    // f32 -> f16 conversions
    k_f32_to_f16<<<gs((long long)N * NDIM), 256, 0, stream>>>(emb_ent, entH, (long long)N * NDIM);
    k_f32_to_f16<<<gs((long long)R * NDIM), 256, 0, stream>>>(emb_rel, relH, (long long)R * NDIM);
    k_f32_to_f16<<<gs(64 * KATT), 256, 0, stream>>>(attn_W, attnWH, 64 * KATT);
    k_f32_to_f16<<<gs(64 * KAGG), 256, 0, stream>>>(aggr_W, aggrWH, 64 * KAGG);

    // degree + mean incoming relation
    k_degree<<<4096, 256, 0, stream>>>(emb_rel, tail_idx, rel_idx, selfrel_f, freq, E);
    k_selfrel_fin<<<gs((long long)N * NDIM), 256, 0, stream>>>(selfrel_f, freq, selfrelH, N);

    // WMMA attention pass
    int ntiles = (M + 15) / 16;
    int blocks = (ntiles + WPB - 1) / WPB;
    k_attn<<<blocks, 128, 0, stream>>>(entH, relH, selfrelH, attnWH, attn_b, attn_vec,
                                       head_idx, tail_idx, rel_idx, raw, mx, E, M);

    // softmax: exp + segment sums (beta normalization fused into k_aggr)
    k_exp<<<4096, 256, 0, stream>>>(tail_idx, raw, mx, sums, E, M);

    // WMMA aggregation + scatter
    k_aggr<<<blocks, 128, 0, stream>>>(entH, relH, selfrelH, aggrWH, aggr_b,
                                       head_idx, tail_idx, rel_idx, raw, sums,
                                       (float*)d_out, E, M);
}
